// P2AffinityAggregation_34479997452839
// MI455X (gfx1250) — compile-verified, matrix operand order checked by reading, not run on hardware
//
#include <hip/hip_runtime.h>
#include <hip/hip_bf16.h>

typedef __attribute__((ext_vector_type(16))) _Float16 v16h;
typedef __attribute__((ext_vector_type(8)))  _Float16 v8h;
typedef __attribute__((ext_vector_type(4)))  _Float16 v4h;
typedef __attribute__((ext_vector_type(8)))  float    v8f;

#define N_EDGES   500000
#define MT        2                  // M-tiles (16 edges each) per wave -> 32 edges/wave
#define N_WTILES  (N_EDGES/(16*MT)) // 15625 waves
#define D_FEAT    128
#define L_FEAT    5
#define V2_C      3.5f
#define BETA_C    10.0f
#define STRIDE    264               // padded LDS row stride (halves): kills m*512B bank pattern
#define MSTEP     (16*STRIDE)       // halves between m-tiles in LDS

// Weight tile offsets in d_ws (units: halves). One B-tile = 32 lanes * 16 halves = 512 halves.
#define OFF_E0 0            // 256x128 : 8 kt * 8 nt  = 64 tiles
#define OFF_E1 (64*512)     // 128x128 : 4*8 = 32
#define OFF_E2 (96*512)
#define OFF_E3 (128*512)    // 128x10  -> padded N=16 : 4*1 = 4
#define OFF_D0 (132*512)    // 5x128   -> padded K=32 : 1*8 = 8
#define OFF_D1 (140*512)
#define OFF_D2 (172*512)
#define OFF_D3 (204*512)    // 128x256 : 4*16 = 64
#define W_TOTAL_BYTES ((size_t)268*512*2)   // 274432

// ---------------------------------------------------------------------------
// Prep: convert f32 row-major W[K][N] into f16 WMMA B-fragments, zero padded.
// B tile (32x16 K x N) lane layout: lane L -> n=L%16, h=L/16;
// element e: k_local = e<8 ? 8h+e : 16+8h+(e-8)
// ---------------------------------------------------------------------------
__global__ void convert_weights_kernel(const float* __restrict__ W, int K, int N,
                                       int ktiles, int ntiles, _Float16* __restrict__ dst) {
    int t = blockIdx.x * blockDim.x + threadIdx.x;
    int total = ktiles * ntiles * 32;
    if (t >= total) return;
    int lane = t & 31;
    int tile = t >> 5;
    int kt = tile % ktiles;
    int nt = tile / ktiles;
    int n16 = lane & 15, h = lane >> 4;
    int n = nt * 16 + n16;
    v16h vals;
#pragma unroll
    for (int e = 0; e < 16; ++e) {
        int k_local = (e < 8) ? (8*h + e) : (16 + 8*h + (e - 8));
        int k = kt * 32 + k_local;
        float w = (k < K && n < N) ? W[(size_t)k * N + n] : 0.0f;
        vals[e] = (_Float16)w;
    }
    *(v16h*)(dst + ((size_t)(nt * ktiles + kt) * 32 + lane) * 16) = vals;
}

// A-fragment load: two 16B chunks (k = 8h+0..7 and 16+8h+0..7 within a 32-col band)
__device__ __forceinline__ v16h load_afrag(const _Float16* p) {
    v8h lo = *(const v8h*)p;
    v8h hi = *(const v8h*)(p + 16);
    v16h a;
#pragma unroll
    for (int j = 0; j < 8; ++j) { a[j] = lo[j]; a[j + 8] = hi[j]; }
    return a;
}

// ---------------------------------------------------------------------------
// Hidden FC layer, 2 M-tiles per wave: each B fragment feeds 2 WMMAs.
// in/out point at m-tile 0; m-tile 1 lives at +MSTEP halves.
// ---------------------------------------------------------------------------
template <int KT, int NTILES>
__device__ __forceinline__ void fc_layer2(const _Float16* __restrict__ Wt,
                                          const float* __restrict__ bias, int Nreal,
                                          const _Float16* in, int colin,
                                          _Float16* out, int colout, int lane) {
    const int n16 = lane & 15, h = lane >> 4;
#pragma unroll
    for (int nt = 0; nt < NTILES; ++nt) {
        v8f c0 = {};
        v8f c1 = {};
#pragma unroll
        for (int kt = 0; kt < KT; ++kt) {
            v16h b = *(const v16h*)(Wt + ((size_t)(nt * KT + kt) * 32 + lane) * 16);
            const _Float16* ap = in + n16 * STRIDE + colin + kt * 32 + 8 * h;
            v16h a0 = load_afrag(ap);
            v16h a1 = load_afrag(ap + MSTEP);
            c0 = __builtin_amdgcn_wmma_f32_16x16x32_f16(false, a0, false, b, (short)0, c0, false, false);
            c1 = __builtin_amdgcn_wmma_f32_16x16x32_f16(false, a1, false, b, (short)0, c1, false, false);
        }
        int n = nt * 16 + n16;
        float bv = (n < Nreal) ? bias[n] : 0.0f;
#pragma unroll
        for (int v = 0; v < 8; ++v) {
            float cv0 = fmaxf(c0[v] + bv, 0.0f);
            float cv1 = fmaxf(c1[v] + bv, 0.0f);
            out[(v + 8 * h) * STRIDE + colout + n]         = (_Float16)cv0;
            out[(v + 8 * h) * STRIDE + colout + n + MSTEP] = (_Float16)cv1;
        }
    }
    asm volatile("" ::: "memory");  // keep LDS producer/consumer order across layers
}

// ---------------------------------------------------------------------------
// Main VAE kernel: 1 wave = 1 block = 32 edges (2 WMMA M-tiles).
// ---------------------------------------------------------------------------
__global__ __launch_bounds__(32)
void vae_edge_kernel(const float* __restrict__ x, const int* __restrict__ row,
                     const int* __restrict__ col, const float* __restrict__ eps,
                     const _Float16* __restrict__ Wt,
                     const float* __restrict__ eb0, const float* __restrict__ eb1,
                     const float* __restrict__ eb2, const float* __restrict__ eb3,
                     const float* __restrict__ db0, const float* __restrict__ db1,
                     const float* __restrict__ db2, const float* __restrict__ db3,
                     float* __restrict__ aff, float* __restrict__ recon_part,
                     float* __restrict__ kl_part) {
    __shared__ __attribute__((aligned(32))) _Float16 pair[MT * 16 * STRIDE];
    __shared__ __attribute__((aligned(32))) _Float16 act [MT * 16 * STRIDE];

    const int lane = threadIdx.x & 31;
    const int tile = blockIdx.x;
    const int m16 = lane & 15, h = lane >> 4;

    // lanes 0..15 hold row[], lanes 16..31 hold col[] for each 16-edge m-tile
    int node[MT];
#pragma unroll
    for (int mt = 0; mt < MT; ++mt) {
        int e = tile * (16 * MT) + mt * 16 + m16;
        node[mt] = (h == 0) ? row[e] : col[e];
    }

    // Stage pair = [x[row] | x[col]] as f16: float4 loads -> packed 4xf16 LDS stores.
#pragma unroll
    for (int mt = 0; mt < MT; ++mt) {
        _Float16* pr = pair + mt * MSTEP;
#pragma unroll 4
        for (int i = 0; i < 32; ++i) {
            int idx4 = lane + 32 * i;        // 0..1023 quads (16 edges x 64 quads)
            int m = idx4 >> 6;               // uniform per i
            int f4 = idx4 & 63;              // quad index within 256-feature row
            int nd = __shfl(node[mt], (f4 < 32) ? m : (16 + m), 32);
            const float4 vv = *(const float4*)(x + (size_t)nd * D_FEAT + ((4 * f4) & 127));
            v4h hv;
            hv[0] = (_Float16)vv.x; hv[1] = (_Float16)vv.y;
            hv[2] = (_Float16)vv.z; hv[3] = (_Float16)vv.w;
            *(v4h*)(pr + m * STRIDE + 4 * f4) = hv;
        }
    }
    asm volatile("" ::: "memory");

    // Encoder hidden layers (act ping-pongs between column halves)
    fc_layer2<8, 8>(Wt + OFF_E0, eb0, 128, pair, 0,   act, 0,   lane);
    fc_layer2<4, 8>(Wt + OFF_E1, eb1, 128, act,  0,   act, 128, lane);
    fc_layer2<4, 8>(Wt + OFF_E2, eb2, 128, act,  128, act, 0,   lane);

    // Encoder head: 128 -> 16 (10 real: mu[0..4], logvar[5..9]); reparameterize.
    {
        v8f c[MT];
#pragma unroll
        for (int mt = 0; mt < MT; ++mt) c[mt] = (v8f){};
#pragma unroll
        for (int kt = 0; kt < 4; ++kt) {
            v16h b = *(const v16h*)(Wt + OFF_E3 + ((size_t)kt * 32 + lane) * 16);
            const _Float16* ap = act + m16 * STRIDE + kt * 32 + 8 * h;
#pragma unroll
            for (int mt = 0; mt < MT; ++mt) {
                v16h a = load_afrag(ap + mt * MSTEP);
                c[mt] = __builtin_amdgcn_wmma_f32_16x16x32_f16(false, a, false, b, (short)0, c[mt], false, false);
            }
        }
        float bv = (m16 < 10) ? eb3[m16] : 0.0f;
        const bool isMu = (m16 < L_FEAT);
        float klacc = 0.0f;
#pragma unroll
        for (int mt = 0; mt < MT; ++mt) {
            float zf[8];
#pragma unroll
            for (int v = 0; v < 8; ++v) {
                float muv = c[mt][v] + bv;                         // valid on lanes n<5
                float lgv = __shfl(muv, h * 16 + m16 + 5, 32);     // logvar from lane n+5
                float stdv = __expf(0.5f * lgv);
                float epsv = 0.0f;
                if (isMu)
                    epsv = eps[(size_t)(tile * (16 * MT) + mt * 16 + v + 8 * h) * L_FEAT + m16];
                zf[v] = muv + epsv * stdv;
                if (isMu)
                    klacc += -0.5f * (1.0f + lgv - muv * muv - __expf(lgv)) * BETA_C;
            }
            // z buffer: 16 x 32 (K padded to 32) at act cols 128..159, zero then scatter z.
            _Float16* zb = act + mt * MSTEP;
            v8h zz = {};
            *(v8h*)(zb + m16 * STRIDE + 128 + 16 * h)     = zz;
            *(v8h*)(zb + m16 * STRIDE + 128 + 16 * h + 8) = zz;
            asm volatile("" ::: "memory");
            if (isMu) {
#pragma unroll
                for (int v = 0; v < 8; ++v)
                    zb[(v + 8 * h) * STRIDE + 128 + m16] = (_Float16)zf[v];
            }
            asm volatile("" ::: "memory");
        }
        klacc += __shfl_xor(klacc, 1);
        klacc += __shfl_xor(klacc, 2);
        klacc += __shfl_xor(klacc, 4);
        klacc += __shfl_xor(klacc, 8);
        klacc += __shfl_xor(klacc, 16);
        if (lane == 0) kl_part[tile] = klacc;
    }

    // Decoder hidden layers
    fc_layer2<1, 8>(Wt + OFF_D0, db0, 128, act, 128, act, 0,   lane);
    fc_layer2<4, 8>(Wt + OFF_D1, db1, 128, act, 0,   act, 128, lane);
    fc_layer2<4, 8>(Wt + OFF_D2, db2, 128, act, 128, act, 0,   lane);

    // Decoder output 128 -> 256 fused with recon loss vs. pair (still in LDS).
    float sq[MT][8];
#pragma unroll
    for (int mt = 0; mt < MT; ++mt)
#pragma unroll
        for (int v = 0; v < 8; ++v) sq[mt][v] = 0.0f;

#pragma unroll
    for (int nt = 0; nt < 16; ++nt) {
        v8f c[MT];
#pragma unroll
        for (int mt = 0; mt < MT; ++mt) c[mt] = (v8f){};
#pragma unroll
        for (int kt = 0; kt < 4; ++kt) {
            v16h b = *(const v16h*)(Wt + OFF_D3 + ((size_t)(nt * 4 + kt) * 32 + lane) * 16);
            const _Float16* ap = act + m16 * STRIDE + kt * 32 + 8 * h;
#pragma unroll
            for (int mt = 0; mt < MT; ++mt) {
                v16h a = load_afrag(ap + mt * MSTEP);
                c[mt] = __builtin_amdgcn_wmma_f32_16x16x32_f16(false, a, false, b, (short)0, c[mt], false, false);
            }
        }
        float bv = db3[nt * 16 + m16];
#pragma unroll
        for (int mt = 0; mt < MT; ++mt) {
#pragma unroll
            for (int v = 0; v < 8; ++v) {
                float rec = c[mt][v] + bv;
                float pv = (float)pair[mt * MSTEP + (v + 8 * h) * STRIDE + nt * 16 + m16];
                float d = rec - pv;
                sq[mt][v] += d * d;
            }
        }
    }
    // reduce across the 16 output columns held by lanes (n = lane%16)
    float rsum = 0.0f;
#pragma unroll
    for (int mt = 0; mt < MT; ++mt) {
#pragma unroll
        for (int v = 0; v < 8; ++v) {
            sq[mt][v] += __shfl_xor(sq[mt][v], 1);
            sq[mt][v] += __shfl_xor(sq[mt][v], 2);
            sq[mt][v] += __shfl_xor(sq[mt][v], 4);
            sq[mt][v] += __shfl_xor(sq[mt][v], 8);
        }
        float rl[8];
#pragma unroll
        for (int v = 0; v < 8; ++v) { rl[v] = sqrtf(sq[mt][v]); rsum += rl[v]; }
        if (m16 == 0) {
#pragma unroll
            for (int v = 0; v < 8; ++v) {
                int e = tile * (16 * MT) + mt * 16 + v + 8 * h;
                aff[e] = 1.0f / (1.0f + V2_C * rl[v]);
            }
        }
    }
    rsum += __shfl_xor(rsum, 16);
    if (lane == 0) recon_part[tile] = rsum;
}

// ---------------------------------------------------------------------------
// Deterministic fixed-order final reduction for the two means.
// ---------------------------------------------------------------------------
__global__ void finalize_kernel(const float* __restrict__ rp, const float* __restrict__ kp,
                                float* __restrict__ means) {
    __shared__ float s1[256];
    __shared__ float s2[256];
    int t = threadIdx.x;
    float a = 0.0f, b = 0.0f;
    for (int i = t; i < N_WTILES; i += 256) { a += rp[i]; b += kp[i]; }
    s1[t] = a; s2[t] = b;
    __syncthreads();
    for (int s = 128; s > 0; s >>= 1) {
        if (t < s) { s1[t] += s1[t + s]; s2[t] += s2[t + s]; }
        __syncthreads();
    }
    if (t == 0) {
        means[0] = s1[0] / (float)N_EDGES;                       // recon_loss.mean()
        means[1] = s2[0] / ((float)N_EDGES * (float)L_FEAT);     // kl_loss.mean()
    }
}

// ---------------------------------------------------------------------------
extern "C" void kernel_launch(void* const* d_in, const int* in_sizes, int n_in,
                              void* d_out, int out_size, void* d_ws, size_t ws_size,
                              hipStream_t stream) {
    const float* x   = (const float*)d_in[0];
    const int*   row = (const int*)d_in[1];
    const int*   col = (const int*)d_in[2];
    const float* eps = (const float*)d_in[3];
    const float* eW0 = (const float*)d_in[4];  const float* eb0 = (const float*)d_in[5];
    const float* eW1 = (const float*)d_in[6];  const float* eb1 = (const float*)d_in[7];
    const float* eW2 = (const float*)d_in[8];  const float* eb2 = (const float*)d_in[9];
    const float* eW3 = (const float*)d_in[10]; const float* eb3 = (const float*)d_in[11];
    const float* dW0 = (const float*)d_in[12]; const float* db0 = (const float*)d_in[13];
    const float* dW1 = (const float*)d_in[14]; const float* db1 = (const float*)d_in[15];
    const float* dW2 = (const float*)d_in[16]; const float* db2 = (const float*)d_in[17];
    const float* dW3 = (const float*)d_in[18]; const float* db3 = (const float*)d_in[19];

    _Float16* Wt = (_Float16*)d_ws;
    float* recon_part = (float*)((char*)d_ws + W_TOTAL_BYTES);
    float* kl_part    = recon_part + N_WTILES;
    float* out        = (float*)d_out;

    struct LayerDesc { const float* W; int K, N, kt, nt; size_t off; };
    const LayerDesc L[8] = {
        { eW0, 256, 128, 8, 8,  OFF_E0 },
        { eW1, 128, 128, 4, 8,  OFF_E1 },
        { eW2, 128, 128, 4, 8,  OFF_E2 },
        { eW3, 128, 10,  4, 1,  OFF_E3 },
        { dW0, 5,   128, 1, 8,  OFF_D0 },
        { dW1, 128, 128, 4, 8,  OFF_D1 },
        { dW2, 128, 128, 4, 8,  OFF_D2 },
        { dW3, 128, 256, 4, 16, OFF_D3 },
    };
    for (int i = 0; i < 8; ++i) {
        int total = L[i].kt * L[i].nt * 32;
        convert_weights_kernel<<<(total + 255) / 256, 256, 0, stream>>>(
            L[i].W, L[i].K, L[i].N, L[i].kt, L[i].nt, Wt + L[i].off);
    }

    vae_edge_kernel<<<N_WTILES, 32, 0, stream>>>(
        x, row, col, eps, Wt, eb0, eb1, eb2, eb3, db0, db1, db2, db3,
        out, recon_part, kl_part);

    finalize_kernel<<<1, 256, 0, stream>>>(recon_part, kl_part, out + N_EDGES);
}
